// GATClassifier_25348896981472
// MI455X (gfx1250) — matrix operand order repaired and checked by reading, not compiled, not run on hardware
//
#include <hip/hip_runtime.h>
#include <hip/hip_bf16.h>
#include <hip/hip_fp16.h>

#define N_NODES 4096
#define N_EDGES 65536
#define E2      (N_EDGES + N_NODES)   /* 69632 edges incl. self loops */
#define IN_CH   1300
#define KP1     1312                  /* IN_CH padded to multiple of 32 */
#define HID     512
#define HEADS   4
#define D1      (HEADS * HID)         /* 2048 */
#define N_GRAPHS 32
#define NEG_SLOPE 0.2f

typedef __attribute__((ext_vector_type(16))) _Float16 v16h;
typedef __attribute__((ext_vector_type(8)))  _Float16 v8h;
typedef __attribute__((ext_vector_type(8)))  float    v8f;

// ---------------------------------------------------------------------------
// fp32 -> f16 convert with K padding:  dst[r, 0..Kp) = src[r, 0..K) | 0
// ---------------------------------------------------------------------------
__global__ void cvt_pad_f16(const float* __restrict__ src, _Float16* __restrict__ dst,
                            int K, int Kp, int total /* rows*Kp */)
{
    int idx = blockIdx.x * blockDim.x + threadIdx.x;
    if (idx >= total) return;
    int r = idx / Kp, k = idx - r * Kp;
    dst[idx] = (k < K) ? (_Float16)src[(size_t)r * K + k] : (_Float16)0.f;
}

// ---------------------------------------------------------------------------
// fp32 W[K,N] -> f16 Wt[N,Kp] (transpose + pad), 32x32 LDS tiles
// grid = (Kp/32, N/32), block = (32,8)
// ---------------------------------------------------------------------------
__global__ __launch_bounds__(256)
void cvt_transpose_f16(const float* __restrict__ W, _Float16* __restrict__ Wt,
                       int K, int N, int Kp)
{
    __shared__ float tile[32][33];
    int kb = blockIdx.x * 32;
    int nb = blockIdx.y * 32;
    #pragma unroll
    for (int i = threadIdx.y; i < 32; i += 8) {
        int k = kb + i;
        tile[i][threadIdx.x] = (k < K) ? W[(size_t)k * N + nb + threadIdx.x] : 0.f;
    }
    __syncthreads();
    #pragma unroll
    for (int i = threadIdx.y; i < 32; i += 8) {
        int n = nb + i, k = kb + threadIdx.x;
        Wt[(size_t)n * Kp + k] = (_Float16)tile[threadIdx.x][i];
    }
}

// ---------------------------------------------------------------------------
// Async global->LDS 16-byte copy (ASYNCcnt-tracked DMA, no VGPR round-trip).
// LDS aperture keeps the LDS offset in flat-addr[31:0].
// ---------------------------------------------------------------------------
__device__ __forceinline__ unsigned ldsOff(const void* p)
{
    return (unsigned)(uintptr_t)p;
}
__device__ __forceinline__ void asyncCopy16(unsigned lds, const _Float16* g)
{
    asm volatile("global_load_async_to_lds_b128 %0, %1, off"
                 :: "v"(lds), "v"(g) : "memory");
}
__device__ __forceinline__ void waitAsync0()
{
    asm volatile("s_wait_asynccnt 0x0" ::: "memory");
}
__device__ __forceinline__ void waitAsync4()
{
    asm volatile("s_wait_asynccnt 0x4" ::: "memory");
}

// ---------------------------------------------------------------------------
// WMMA GEMM: C[M,N] = A[M,Kp] @ Bt[N,Kp]^T, f16 in, fp32 out. Kp % 32 == 0.
// Block tile 128x128, 8 waves (4x2), each wave 32x64 = 2x4 WMMA 16x16x32.
// 3-deep LDS ring buffer fed by async global->LDS b128 copies issued two
// tiles ahead; uniform partial-count asynccnt waits (async loads complete in
// order). Tail iterations issue dummy copies into the dead ring slot so the
// loop is branch-free; s_endpgm's implicit wait-idle drains them.
// ---------------------------------------------------------------------------
#define BM 128
#define BN 128
#define KB 32
#define AST 40   /* padded LDS stride (halves): 80B rows -> 16B aligned, conflict-free */
#define BST 40

__global__ __launch_bounds__(256)
void gemm_wmma_f16(const _Float16* __restrict__ A,   // [M, Kp] row-major, padded
                   const _Float16* __restrict__ Bt,  // [N, Kp] row-major (= B^T)
                   float* __restrict__ C, int M, int N, int Kp)
{
    __shared__ _Float16 As[3][BM * AST];
    __shared__ _Float16 Bs[3][BN * BST];

    const int t    = threadIdx.x;
    const int lane = t & 31;
    const int wave = t >> 5;
    const int wm   = wave >> 1;      // 0..3  (row band of 32)
    const int wn   = wave & 1;       // 0..1  (col band of 64)
    const int hi   = lane >> 4;      // 0..1  (lane half)
    const int l16  = lane & 15;
    const int bm   = blockIdx.y * BM;
    const int bn   = blockIdx.x * BN;

    // per-thread 16B copy slots: 512 chunks per operand tile, 2 per thread
    const int crow = t >> 2;                 // 0..63  (+64 for second chunk)
    const int cc8  = (t & 3) * 8;
    const _Float16* aSrc0 = A  + (size_t)(bm + crow) * Kp + cc8;
    const _Float16* aSrc1 = aSrc0 + (size_t)64 * Kp;
    const _Float16* bSrc0 = Bt + (size_t)(bn + crow) * Kp + cc8;
    const _Float16* bSrc1 = bSrc0 + (size_t)64 * Kp;
    unsigned aD0[3], aD1[3], bD0[3], bD1[3];
    #pragma unroll
    for (int b = 0; b < 3; ++b) {
        aD0[b] = ldsOff(&As[b][crow * AST + cc8]);
        aD1[b] = ldsOff(&As[b][(crow + 64) * AST + cc8]);
        bD0[b] = ldsOff(&Bs[b][crow * BST + cc8]);
        bD1[b] = ldsOff(&Bs[b][(crow + 64) * BST + cc8]);
    }

    v8f acc[2][4] = {};

    const int ksteps = Kp / KB;

    // prologue: stage tiles 0 and 1 (8 copies in flight, in-order completion)
    asyncCopy16(aD0[0], aSrc0);  asyncCopy16(aD1[0], aSrc1);
    asyncCopy16(bD0[0], bSrc0);  asyncCopy16(bD1[0], bSrc1);
    if (ksteps > 1) {
        asyncCopy16(aD0[1], aSrc0 + KB);  asyncCopy16(aD1[1], aSrc1 + KB);
        asyncCopy16(bD0[1], bSrc0 + KB);  asyncCopy16(bD1[1], bSrc1 + KB);
        waitAsync4();                                       // tile 0 resident
    } else {
        waitAsync0();
    }
    __syncthreads();

    int cur = 0, nx2 = 2;
    for (int ks = 0; ks < ksteps; ++ks) {
        // issue copies for tile ks+2 into ring slot nx2; in the last two
        // iterations this re-copies the final tile into a dead slot (dummy)
        // so control flow and the wait threshold stay uniform.
        const int kpre = (ks + 2 < ksteps) ? (ks + 2) : (ksteps - 1);
        const int ko   = kpre * KB;
        asyncCopy16(aD0[nx2], aSrc0 + ko);  asyncCopy16(aD1[nx2], aSrc1 + ko);
        asyncCopy16(bD0[nx2], bSrc0 + ko);  asyncCopy16(bD1[nx2], bSrc1 + ko);

        // ---- A fragments: documented 16-bit A layout
        // lane half 0: VGPR0-3 = K 0..7,  VGPR4-7 = K 16..23
        // lane half 1: VGPR0-3 = K 8..15, VGPR4-7 = K 24..31
        v16h afrag[2], bfrag[4];
        #pragma unroll
        for (int tm = 0; tm < 2; ++tm) {
            int m = wm * 32 + tm * 16 + l16;
            v8h lo = *(const v8h*)&As[cur][m * AST + hi * 8];
            v8h hh = *(const v8h*)&As[cur][m * AST + 16 + hi * 8];
            #pragma unroll
            for (int j = 0; j < 8; ++j) { afrag[tm][j] = lo[j]; afrag[tm][8 + j] = hh[j]; }
        }
        // ---- B fragments: same K layout as A (Bs holds B^T)
        #pragma unroll
        for (int tn = 0; tn < 4; ++tn) {
            int n = wn * 64 + tn * 16 + l16;
            v8h lo = *(const v8h*)&Bs[cur][n * BST + hi * 8];
            v8h hh = *(const v8h*)&Bs[cur][n * BST + 16 + hi * 8];
            #pragma unroll
            for (int j = 0; j < 8; ++j) { bfrag[tn][j] = lo[j]; bfrag[tn][8 + j] = hh[j]; }
        }

        #pragma unroll
        for (int tm = 0; tm < 2; ++tm)
            #pragma unroll
            for (int tn = 0; tn < 4; ++tn)
                acc[tm][tn] = __builtin_amdgcn_wmma_f32_16x16x32_f16(
                    false, afrag[tm], false, bfrag[tn],
                    (short)0, acc[tm][tn], false, false);

        // scheduling fence keeps the wait AFTER the WMMA block (no register
        // copies needed), so the in-flight DMA overlaps the compute phase;
        // wait <=4 leaves the newest 4 copies (tile ks+2) in flight while
        // guaranteeing tile ks+1 is resident (in-order completion).
        __builtin_amdgcn_sched_barrier(0);
        waitAsync4();
        __syncthreads();

        cur = (cur == 2) ? 0 : cur + 1;
        nx2 = (nx2 == 2) ? 0 : nx2 + 1;
    }

    // ---- store C (f32 C/D layout: VGPR v = row hi*8+v, col l16) ----
    #pragma unroll
    for (int tm = 0; tm < 2; ++tm) {
        #pragma unroll
        for (int tn = 0; tn < 4; ++tn) {
            int n = bn + wn * 64 + tn * 16 + l16;
            #pragma unroll
            for (int v = 0; v < 8; ++v) {
                int m = bm + wm * 32 + tm * 16 + hi * 8 + v;
                C[(size_t)m * N + n] = acc[tm][tn][v];
            }
        }
    }
}

// ---------------------------------------------------------------------------
// Attention-score reduction: a_src[n,h] = h[n,h,:].att_src[h,:] (wave per head)
// ---------------------------------------------------------------------------
__global__ __launch_bounds__(128)
void attn_scores(const float* __restrict__ h, const float* __restrict__ att_src,
                 const float* __restrict__ att_dst,
                 float* __restrict__ a_src, float* __restrict__ a_dst)
{
    int node = blockIdx.x;
    int head = threadIdx.x >> 5;
    int lane = threadIdx.x & 31;
    const float* hp = h + (size_t)node * D1 + head * HID;
    const float* as = att_src + head * HID;
    const float* ad = att_dst + head * HID;
    float s = 0.f, d = 0.f;
    #pragma unroll 4
    for (int c = lane; c < HID; c += 32) {
        float v = hp[c];
        s += v * as[c];
        d += v * ad[c];
    }
    #pragma unroll
    for (int off = 16; off > 0; off >>= 1) {
        s += __shfl_xor(s, off, 32);
        d += __shfl_xor(d, off, 32);
    }
    if (lane == 0) {
        a_src[node * HEADS + head] = s;
        a_dst[node * HEADS + head] = d;
    }
}

__device__ __forceinline__ void edge_sd(const int* __restrict__ ei, int e, int& s, int& d)
{
    if (e < N_EDGES) { s = ei[e]; d = ei[N_EDGES + e]; }
    else             { s = e - N_EDGES; d = s; }        // self loop
}

__device__ __forceinline__ void atomicMaxF(float* addr, float val)
{
    // monotone bit trick (buffer initialized to -inf)
    if (val >= 0.f) atomicMax((int*)addr, __float_as_int(val));
    else            atomicMin((unsigned int*)addr, __float_as_uint(val));
}

__global__ void edge_score_max(const int* __restrict__ ei,
                               const float* __restrict__ a_src,
                               const float* __restrict__ a_dst,
                               float* __restrict__ escore, float* __restrict__ emax)
{
    int idx = blockIdx.x * blockDim.x + threadIdx.x;
    if (idx >= E2 * HEADS) return;
    int e = idx >> 2, h = idx & 3;
    int s, d; edge_sd(ei, e, s, d);
    float v = a_src[s * HEADS + h] + a_dst[d * HEADS + h];
    v = (v > 0.f) ? v : NEG_SLOPE * v;                 // leaky relu
    escore[idx] = v;
    atomicMaxF(&emax[d * HEADS + h], v);
}

__global__ void edge_exp_sum(const int* __restrict__ ei,
                             float* __restrict__ escore,
                             const float* __restrict__ emax,
                             float* __restrict__ denom)
{
    int idx = blockIdx.x * blockDim.x + threadIdx.x;
    if (idx >= E2 * HEADS) return;
    int e = idx >> 2, h = idx & 3;
    int s, d; edge_sd(ei, e, s, d);
    float ex = __expf(escore[idx] - emax[d * HEADS + h]);
    escore[idx] = ex;
    unsafeAtomicAdd(&denom[d * HEADS + h], ex);        // global_atomic_add_f32
}

__global__ __launch_bounds__(256)
void edge_aggregate(const int* __restrict__ ei, const float* __restrict__ hfeat,
                    const float* __restrict__ escore, const float* __restrict__ denom,
                    float* __restrict__ agg)
{
    __shared__ float alpha[HEADS];
    __shared__ int sS, sD;
    int e = blockIdx.x;
    if (threadIdx.x == 0) { int s, d; edge_sd(ei, e, s, d); sS = s; sD = d; }
    __syncthreads();
    int s = sS, d = sD;
    if (threadIdx.x < HEADS)
        alpha[threadIdx.x] = escore[e * HEADS + threadIdx.x]
                           / (denom[d * HEADS + threadIdx.x] + 1e-16f);
    __syncthreads();
    const float* hp = hfeat + (size_t)s * D1;
    float*       op = agg   + (size_t)d * D1;
    #pragma unroll
    for (int i = 0; i < D1 / 256; ++i) {
        int idx = i * 256 + threadIdx.x;
        float w = alpha[idx >> 9];                     // HID = 512
        unsafeAtomicAdd(&op[idx], hp[idx] * w);
    }
}

// bias + relu in place, also emit f16 copy (feeds next-layer GEMM directly)
__global__ void bias_relu_cvt(float* __restrict__ x, const float* __restrict__ bias,
                              _Float16* __restrict__ xh)
{
    int idx = blockIdx.x * blockDim.x + threadIdx.x;
    if (idx >= N_NODES * D1) return;
    float v = x[idx] + bias[idx & (D1 - 1)];
    v = v > 0.f ? v : 0.f;
    x[idx]  = v;
    xh[idx] = (_Float16)v;
}

__global__ void bias_relu(float* __restrict__ x, const float* __restrict__ bias)
{
    int idx = blockIdx.x * blockDim.x + threadIdx.x;
    if (idx >= N_NODES * D1) return;
    float v = x[idx] + bias[idx & (D1 - 1)];
    x[idx] = v > 0.f ? v : 0.f;
}

__global__ void fill_f32(float* __restrict__ p, float val, int n)
{
    int i = blockIdx.x * blockDim.x + threadIdx.x;
    if (i < n) p[i] = val;
}

__global__ __launch_bounds__(256)
void node_dot_pool(const float* __restrict__ h, const float* __restrict__ linW,
                   const int* __restrict__ batch,
                   float* __restrict__ gsum, float* __restrict__ gcnt)
{
    __shared__ float part[8];
    int node = blockIdx.x;
    int lane = threadIdx.x & 31, wave = threadIdx.x >> 5;
    const float* hp = h + (size_t)node * D1;
    float s = 0.f;
    for (int i = threadIdx.x; i < D1; i += 256) s += hp[i] * linW[i];
    #pragma unroll
    for (int off = 16; off > 0; off >>= 1) s += __shfl_xor(s, off, 32);
    if (lane == 0) part[wave] = s;
    __syncthreads();
    if (threadIdx.x == 0) {
        float tot = 0.f;
        #pragma unroll
        for (int w = 0; w < 8; ++w) tot += part[w];
        int g = batch[node];
        unsafeAtomicAdd(&gsum[g], tot);
        unsafeAtomicAdd(&gcnt[g], 1.0f);
    }
}

__global__ void finalize(const float* __restrict__ gsum, const float* __restrict__ gcnt,
                         const float* __restrict__ lin_b, float* __restrict__ out)
{
    int g = threadIdx.x;
    if (g < N_GRAPHS) out[g] = gsum[g] / fmaxf(gcnt[g], 1.0f) + lin_b[0];
}

// ---------------------------------------------------------------------------
extern "C" void kernel_launch(void* const* d_in, const int* in_sizes, int n_in,
                              void* d_out, int out_size, void* d_ws, size_t ws_size,
                              hipStream_t stream)
{
    const float* x      = (const float*)d_in[0];
    const int*   ei     = (const int*)  d_in[1];
    const int*   batch  = (const int*)  d_in[2];
    const float* W1     = (const float*)d_in[3];
    const float* att_s1 = (const float*)d_in[4];
    const float* att_d1 = (const float*)d_in[5];
    const float* b1     = (const float*)d_in[6];
    const float* W2     = (const float*)d_in[7];
    const float* att_s2 = (const float*)d_in[8];
    const float* att_d2 = (const float*)d_in[9];
    const float* b2     = (const float*)d_in[10];
    const float* lin_W  = (const float*)d_in[11];
    const float* lin_b  = (const float*)d_in[12];
    float*       out    = (float*)d_out;

    // ---- workspace layout (all offsets multiples of 16B) ----
    float* ws     = (float*)d_ws;
    float* h1     = ws;                                  // 4096*2048 f32
    float* agg    = h1   + (size_t)N_NODES * D1;         // 4096*2048 f32
    float* a_src  = agg  + (size_t)N_NODES * D1;         // 4096*4
    float* a_dst  = a_src + N_NODES * HEADS;
    float* emax   = a_dst + N_NODES * HEADS;
    float* denom  = emax  + N_NODES * HEADS;
    float* escore = denom + N_NODES * HEADS;             // 69632*4
    float* gsum   = escore + (size_t)E2 * HEADS;         // 32
    float* gcnt   = gsum + N_GRAPHS;                     // 32
    _Float16* Ah  = (_Float16*)(gcnt + N_GRAPHS);        // max 4096*2048 f16
    _Float16* Bth = Ah + (size_t)N_NODES * D1;           // max 2048*2048 f16 (B^T)

    const dim3 gemm_grid(D1 / BN, N_NODES / BM);         // (16, 32)
    const int nh   = N_NODES * HEADS;
    const int eh   = E2 * HEADS;
    const int feat = N_NODES * D1;
    const float NEG_INF = -__builtin_huge_valf();

    // ---------------- layer 1 ----------------
    {
        const int totA = N_NODES * KP1;
        cvt_pad_f16<<<(totA + 255) / 256, 256, 0, stream>>>(x, Ah, IN_CH, KP1, totA);
        cvt_transpose_f16<<<dim3(KP1 / 32, D1 / 32), dim3(32, 8), 0, stream>>>(
            W1, Bth, IN_CH, D1, KP1);
        gemm_wmma_f16<<<gemm_grid, 256, 0, stream>>>(Ah, Bth, h1, N_NODES, D1, KP1);
    }
    attn_scores<<<N_NODES, 128, 0, stream>>>(h1, att_s1, att_d1, a_src, a_dst);
    fill_f32<<<(nh + 255) / 256, 256, 0, stream>>>(emax, NEG_INF, nh);
    fill_f32<<<(nh + 255) / 256, 256, 0, stream>>>(denom, 0.f, nh);
    fill_f32<<<(feat + 255) / 256, 256, 0, stream>>>(agg, 0.f, feat);
    edge_score_max<<<(eh + 255) / 256, 256, 0, stream>>>(ei, a_src, a_dst, escore, emax);
    edge_exp_sum<<<(eh + 255) / 256, 256, 0, stream>>>(ei, escore, emax, denom);
    edge_aggregate<<<E2, 256, 0, stream>>>(ei, h1, escore, denom, agg);
    // bias+relu in place AND emit f16 feed for the next GEMM (D1 % 32 == 0)
    bias_relu_cvt<<<(feat + 255) / 256, 256, 0, stream>>>(agg, b1, Ah);

    // ---------------- layer 2 (agg -> h1 -> agg) ----------------
    {
        cvt_transpose_f16<<<dim3(D1 / 32, D1 / 32), dim3(32, 8), 0, stream>>>(
            W2, Bth, D1, D1, D1);
        gemm_wmma_f16<<<gemm_grid, 256, 0, stream>>>(Ah, Bth, h1, N_NODES, D1, D1);
    }
    attn_scores<<<N_NODES, 128, 0, stream>>>(h1, att_s2, att_d2, a_src, a_dst);
    fill_f32<<<(nh + 255) / 256, 256, 0, stream>>>(emax, NEG_INF, nh);
    fill_f32<<<(nh + 255) / 256, 256, 0, stream>>>(denom, 0.f, nh);
    fill_f32<<<(feat + 255) / 256, 256, 0, stream>>>(agg, 0.f, feat);
    edge_score_max<<<(eh + 255) / 256, 256, 0, stream>>>(ei, a_src, a_dst, escore, emax);
    edge_exp_sum<<<(eh + 255) / 256, 256, 0, stream>>>(ei, escore, emax, denom);
    edge_aggregate<<<E2, 256, 0, stream>>>(ei, h1, escore, denom, agg);
    bias_relu<<<(feat + 255) / 256, 256, 0, stream>>>(agg, b2);

    // ---------------- pool + linear ----------------
    fill_f32<<<1, 64, 0, stream>>>(gsum, 0.f, 2 * N_GRAPHS);   // gsum+gcnt contiguous
    node_dot_pool<<<N_NODES, 256, 0, stream>>>(agg, lin_W, batch, gsum, gcnt);
    finalize<<<1, N_GRAPHS, 0, stream>>>(gsum, gcnt, lin_b, out);
}
